// BertLayer_19756849562189
// MI455X (gfx1250) — matrix operand order repaired and checked by reading, not compiled, not run on hardware
//
#include <hip/hip_runtime.h>
#include <stdint.h>

// ---------------------------------------------------------------------------
// BERT layer (B=2, S=2048, H=768, NH=12, DH=64, FF=3072, AD=64) for gfx1250.
// f16 operands + v_wmma_f32_16x16x32_f16; TDM (tensor_load_to_lds) staging
// with double-buffered LDS tiles and s_wait_tensorcnt synchronization.
// ---------------------------------------------------------------------------

typedef __attribute__((ext_vector_type(16))) _Float16     v16h;
typedef __attribute__((ext_vector_type(8)))  float        v8f;
typedef __attribute__((ext_vector_type(4)))  unsigned int u32x4;
typedef __attribute__((ext_vector_type(8)))  int          i32x8;
typedef __attribute__((ext_vector_type(4)))  int          i32x4;

#define WMMA_F32_F16(A, B, C) \
    __builtin_amdgcn_wmma_f32_16x16x32_f16(false, (A), false, (B), (short)0, (C), false, false)

__device__ __forceinline__ float gelu_exact(float x) {
    return 0.5f * x * (1.0f + erff(x * 0.70710678118654752f));
}

// ---------------------------------------------------------------------------
// Tensor Data Mover: 2-D tile (f16 elements) global -> LDS, with LDS row
// padding. D# layout per CDNA5 ISA 8.3/8.4: group0 = {flags, lds_addr,
// global_addr}, group1 = {data_size/pad, tensor dims, tile dims, strides}.
// pad_interval_code: pad every 2^(c+1) DWORDs; pad_amount_code: c+1 DWORDs.
// Issue from one wave only (TDM ignores EXEC).
// ---------------------------------------------------------------------------
__device__ __forceinline__ void tdm_load_2d(
    unsigned lds_byte_off, const void* gptr,
    unsigned tile_d0 /*elems*/, unsigned tile_d1 /*rows*/,
    unsigned tensor_d0, unsigned tensor_d1,
    unsigned stride_elems,
    unsigned pad_interval_code, unsigned pad_amount_code)
{
    unsigned long long ga = (unsigned long long)(uintptr_t)gptr;
    u32x4 g0;
    g0[0] = 1u;                                        // count=1, user-mode D#
    g0[1] = lds_byte_off;                              // lds_addr
    g0[2] = (unsigned)ga;                              // global_addr[31:0]
    g0[3] = (unsigned)((ga >> 32) & 0x01ffffffu) | (2u << 30);  // addr[56:32] | type=2
    i32x8 g1;
    g1[0] = (int)(0x10000u                             // data_size=1 (2 bytes)
                  | (1u << 20)                         // pad_enable
                  | (pad_interval_code << 22)
                  | (pad_amount_code << 25));
    g1[1] = (int)((tensor_d0 & 0xffffu) << 16);                          // dim0[15:0]
    g1[2] = (int)(((tensor_d0 >> 16) & 0xffffu) | ((tensor_d1 & 0xffffu) << 16));
    g1[3] = (int)(((tensor_d1 >> 16) & 0xffffu) | ((tile_d0 & 0xffffu) << 16));
    g1[4] = (int)(tile_d1 & 0xffffu);                  // tile_dim1, tile_dim2=0
    g1[5] = (int)stride_elems;                         // tensor_dim0_stride[31:0]
    g1[6] = 0;                                         // stride[47:32], dim1_stride lo
    g1[7] = 0;
    i32x4 z4 = {0, 0, 0, 0};
#if __clang_major__ >= 23
    i32x8 z8 = {0, 0, 0, 0, 0, 0, 0, 0};
    __builtin_amdgcn_tensor_load_to_lds(g0, g1, z4, z4, z8, 0);
#else
    __builtin_amdgcn_tensor_load_to_lds(g0, g1, z4, z4, 0);
#endif
}

// ---------------------------------------------------------------------------
// fp32 -> f16 conversion pass (vectorized 16B loads, 8B stores).
// ---------------------------------------------------------------------------
__global__ __launch_bounds__(256) void cvt_kernel(
    const float* __restrict__ s, _Float16* __restrict__ d, int n)
{
    int i = (blockIdx.x * 256 + threadIdx.x) * 4;
    if (i < n) {
        float4 v = *(const float4*)(s + i);
        d[i + 0] = (_Float16)v.x;
        d[i + 1] = (_Float16)v.y;
        d[i + 2] = (_Float16)v.z;
        d[i + 3] = (_Float16)v.w;
    }
}

// ---------------------------------------------------------------------------
// GEMM: out = act(A[M,K] @ W[N,K]^T + bias) (+ residual). A/W are f16.
// Block tile 128x64, K-step 32, 8 waves x (16x64). TDM double-buffered LDS.
// Optional f32 and f16 outputs (f16 feeds the next GEMM's A operand).
// ---------------------------------------------------------------------------
__global__ __launch_bounds__(256) void gemm_f16(
    const _Float16* __restrict__ A, const _Float16* __restrict__ W,
    const float* __restrict__ bias, const float* __restrict__ residual,
    float* __restrict__ Cf, _Float16* __restrict__ Ch,
    int M, int N, int K, int act)
{
    constexpr int BM = 128, BN = 64, BK = 32;
    constexpr int SA = BK + 8, SB = BK + 8;           // 40-half padded rows (TDM pad)
    __shared__ _Float16 sA[2 * BM * SA];
    __shared__ _Float16 sB[2 * BN * SB];

    const int tid  = threadIdx.x;
    const int lane = tid & 31;
    const int wave = tid >> 5;
    const int m0   = blockIdx.y * BM;
    const int n0   = blockIdx.x * BN;

    const int mm      = lane & 15;
    const int kofs8   = (lane >> 4) << 3;
    const int kbase16 = (lane >> 4) << 4;
    const int rofs    = (lane >> 4) << 3;

    // flat LDS address truncated to 32 bits == LDS byte offset (ISA 10.2)
    const unsigned sAoff = (unsigned)(uintptr_t)(void*)sA;
    const unsigned sBoff = (unsigned)(uintptr_t)(void*)sB;

    // prime buffer 0: 32x{128,64} f16 tiles, pad 4 DWORDs every 16 DWORDs -> stride 40
    if (tid < 32) {
        tdm_load_2d(sAoff, A + (size_t)m0 * K, BK, BM, (unsigned)K, (unsigned)M, (unsigned)K, 3, 3);
        tdm_load_2d(sBoff, W + (size_t)n0 * K, BK, BN, (unsigned)K, (unsigned)N, (unsigned)K, 3, 3);
    }

    v8f acc[4] = {};
    int buf = 0;
    for (int k0 = 0; k0 < K; k0 += BK, buf ^= 1) {
        if (tid < 32) __builtin_amdgcn_s_wait_tensorcnt(0);
        __syncthreads();                              // tile[buf] ready; prev reads done
        if (tid < 32 && (k0 + BK) < K) {              // overlap next DMA with compute
            tdm_load_2d(sAoff + (unsigned)((buf ^ 1) * BM * SA * 2),
                        A + (size_t)m0 * K + (k0 + BK), BK, BM, (unsigned)K, (unsigned)M, (unsigned)K, 3, 3);
            tdm_load_2d(sBoff + (unsigned)((buf ^ 1) * BN * SB * 2),
                        W + (size_t)n0 * K + (k0 + BK), BK, BN, (unsigned)K, (unsigned)N, (unsigned)K, 3, 3);
        }
        const _Float16* cA = sA + buf * BM * SA;
        const _Float16* cB = sB + buf * BN * SB;

        v16h afrag;                                   // ISA 16-bit A(16x32) layout
        #pragma unroll
        for (int i = 0; i < 16; ++i) {
            int kk = ((i >> 3) << 4) + kofs8 + (i & 7);
            afrag[i] = cA[(wave * 16 + mm) * SA + kk];
        }
        #pragma unroll
        for (int nt = 0; nt < 4; ++nt) {
            v16h bfrag;                               // B(32x16): [k][n] = W[n][k]
            #pragma unroll
            for (int i = 0; i < 16; ++i)
                bfrag[i] = cB[(nt * 16 + mm) * SB + ((i >> 3) << 4) + kofs8 + (i & 7)];
            acc[nt] = WMMA_F32_F16(afrag, bfrag, acc[nt]);
        }
    }

    #pragma unroll
    for (int nt = 0; nt < 4; ++nt) {
        int n = n0 + nt * 16 + mm;
        float bv = bias ? bias[n] : 0.0f;
        #pragma unroll
        for (int r = 0; r < 8; ++r) {
            int m = m0 + wave * 16 + rofs + r;
            float v = acc[nt][r] + bv;
            if (act == 1) v = gelu_exact(v);
            if (residual) v += residual[(size_t)m * N + n];
            if (Cf) Cf[(size_t)m * N + n] = v;
            if (Ch) Ch[(size_t)m * N + n] = (_Float16)v;
        }
    }
}

// ---------------------------------------------------------------------------
// Flash attention: grid (S/128, NH, B); 8 waves x 16 queries; K/V chunks of 32
// keys TDM-streamed into double-buffered LDS; online softmax; WMMA both GEMMs.
// ---------------------------------------------------------------------------
__global__ __launch_bounds__(256) void attn_kernel(
    const _Float16* __restrict__ Qb, const _Float16* __restrict__ Kb,
    const _Float16* __restrict__ Vb, const float* __restrict__ mask,
    float* __restrict__ Of, _Float16* __restrict__ Oh)
{
    constexpr int S = 2048, H = 768, DH = 64;
    constexpr int SKV = DH + 8;                       // 72-half padded rows (TDM pad)
    constexpr int SP  = 32 + 8;
    __shared__ _Float16 sK[2 * 32 * SKV];
    __shared__ _Float16 sV[2 * 32 * SKV];
    __shared__ _Float16 sP[8 * 16 * SP];

    const int tid  = threadIdx.x;
    const int lane = tid & 31;
    const int wave = tid >> 5;
    const int b    = blockIdx.z;
    const int h    = blockIdx.y;
    const int q0   = blockIdx.x * 128 + wave * 16;

    const int mm      = lane & 15;
    const int kofs8   = (lane >> 4) << 3;
    const int kbase16 = (lane >> 4) << 4;
    const int rofs    = (lane >> 4) << 3;

    const unsigned sKoff = (unsigned)(uintptr_t)(void*)sK;
    const unsigned sVoff = (unsigned)(uintptr_t)(void*)sV;

    // Q fragments (A layout), pre-scaled by 1/sqrt(DH)
    v16h qa[2];
    const _Float16* qrow = Qb + (size_t)(b * S + q0 + mm) * H + h * DH;
    #pragma unroll
    for (int c = 0; c < 2; ++c)
        #pragma unroll
        for (int i = 0; i < 16; ++i) {
            int kk = ((i >> 3) << 4) + kofs8 + (i & 7) + c * 32;
            qa[c][i] = (_Float16)((float)qrow[kk] * 0.125f);
        }

    // prime buffer 0: 32 rows x 64 halves, pad 4 DWORDs every 32 DWORDs -> stride 72
    if (tid < 32) {
        const _Float16* kp = Kb + (size_t)(b * S) * H + h * DH;
        const _Float16* vp = Vb + (size_t)(b * S) * H + h * DH;
        tdm_load_2d(sKoff, kp, DH, 32, DH, 32, H, 4, 3);
        tdm_load_2d(sVoff, vp, DH, 32, DH, 32, H, 4, 3);
    }

    v8f ctx[4] = {};
    float mrow[8], lsum[8];
    #pragma unroll
    for (int r = 0; r < 8; ++r) { mrow[r] = -1e30f; lsum[r] = 0.0f; }

    int buf = 0;
    for (int key0 = 0; key0 < S; key0 += 32, buf ^= 1) {
        if (tid < 32) __builtin_amdgcn_s_wait_tensorcnt(0);
        __syncthreads();
        if (tid < 32 && (key0 + 32) < S) {
            const _Float16* kp = Kb + (size_t)(b * S + key0 + 32) * H + h * DH;
            const _Float16* vp = Vb + (size_t)(b * S + key0 + 32) * H + h * DH;
            tdm_load_2d(sKoff + (unsigned)((buf ^ 1) * 32 * SKV * 2), kp, DH, 32, DH, 32, H, 4, 3);
            tdm_load_2d(sVoff + (unsigned)((buf ^ 1) * 32 * SKV * 2), vp, DH, 32, DH, 32, H, 4, 3);
        }
        const _Float16* cK = sK + buf * 32 * SKV;
        const _Float16* cV = sV + buf * 32 * SKV;

        // scores: 16 queries x 32 keys
        v8f sc[2] = {};
        #pragma unroll
        for (int sub = 0; sub < 2; ++sub)
            #pragma unroll
            for (int c = 0; c < 2; ++c) {
                v16h bfrag;                           // [kd][n] = K[sub*16+n][c*32+kd]
                #pragma unroll
                for (int i = 0; i < 16; ++i)
                    bfrag[i] = cK[(sub * 16 + mm) * SKV + c * 32 + ((i >> 3) << 4) + kofs8 + (i & 7)];
                sc[sub] = WMMA_F32_F16(qa[c], bfrag, sc[sub]);
            }

        // additive mask + online softmax (row reductions over 16-lane groups)
        float mk0 = mask[b * S + key0 + mm];
        float mk1 = mask[b * S + key0 + 16 + mm];
        #pragma unroll
        for (int r = 0; r < 8; ++r) {
            float x0 = sc[0][r] + mk0;
            float x1 = sc[1][r] + mk1;
            float lm = fmaxf(x0, x1);
            #pragma unroll
            for (int off = 1; off < 16; off <<= 1)
                lm = fmaxf(lm, __shfl_xor(lm, off, 32));
            float mnew  = fmaxf(mrow[r], lm);
            float scale = __expf(mrow[r] - mnew);
            float p0 = __expf(x0 - mnew);
            float p1 = __expf(x1 - mnew);
            float rs = p0 + p1;
            #pragma unroll
            for (int off = 1; off < 16; off <<= 1)
                rs += __shfl_xor(rs, off, 32);
            lsum[r] = lsum[r] * scale + rs;
            mrow[r] = mnew;
            #pragma unroll
            for (int nt = 0; nt < 4; ++nt) ctx[nt][r] = ctx[nt][r] * scale;
            sP[(wave * 16 + rofs + r) * SP + mm]      = (_Float16)p0;
            sP[(wave * 16 + rofs + r) * SP + 16 + mm] = (_Float16)p1;
        }
        __syncthreads();                              // P C-layout -> A-layout via LDS

        v16h pfrag;
        #pragma unroll
        for (int i = 0; i < 16; ++i) {
            int kk = ((i >> 3) << 4) + kofs8 + (i & 7);
            pfrag[i] = sP[(wave * 16 + mm) * SP + kk];
        }
        #pragma unroll
        for (int nt = 0; nt < 4; ++nt) {
            v16h vfrag;                               // [k][n] = V[key0+k][nt*16+n]
            #pragma unroll
            for (int i = 0; i < 16; ++i)
                vfrag[i] = cV[(((i >> 3) << 4) + kofs8 + (i & 7)) * SKV + nt * 16 + mm];
            ctx[nt] = WMMA_F32_F16(pfrag, vfrag, ctx[nt]);
        }
    }

    #pragma unroll
    for (int nt = 0; nt < 4; ++nt)
        #pragma unroll
        for (int r = 0; r < 8; ++r) {
            size_t idx = ((size_t)(b * S + q0 + rofs + r)) * H + h * DH + nt * 16 + mm;
            float v = ctx[nt][r] / lsum[r];
            Of[idx] = v;
            Oh[idx] = (_Float16)v;
        }
}

// ---------------------------------------------------------------------------
// LayerNorm over H=768 per row.
// ---------------------------------------------------------------------------
__global__ __launch_bounds__(256) void ln_kernel(
    const float* __restrict__ Y, const float* __restrict__ g,
    const float* __restrict__ bta, float* __restrict__ out)
{
    constexpr int H = 768;
    __shared__ float red1[8], red2[8];
    const int row  = blockIdx.x;
    const int tid  = threadIdx.x;
    const int lane = tid & 31;
    const int wave = tid >> 5;
    const float* yr = Y + (size_t)row * H;

    float s = 0.0f, s2 = 0.0f;
    for (int i = tid; i < H; i += 256) { float v = yr[i]; s += v; s2 += v * v; }
    #pragma unroll
    for (int off = 1; off < 32; off <<= 1) {
        s  += __shfl_xor(s,  off, 32);
        s2 += __shfl_xor(s2, off, 32);
    }
    if (lane == 0) { red1[wave] = s; red2[wave] = s2; }
    __syncthreads();
    if (wave == 0) {
        s  = (lane < 8) ? red1[lane] : 0.0f;
        s2 = (lane < 8) ? red2[lane] : 0.0f;
        #pragma unroll
        for (int off = 1; off < 8; off <<= 1) {
            s  += __shfl_xor(s,  off, 32);
            s2 += __shfl_xor(s2, off, 32);
        }
        if (lane == 0) { red1[0] = s; red2[0] = s2; }
    }
    __syncthreads();
    const float mu   = red1[0] * (1.0f / H);
    const float var  = red2[0] * (1.0f / H) - mu * mu;
    const float rstd = rsqrtf(var + 1e-12f);
    for (int i = tid; i < H; i += 256)
        out[(size_t)row * H + i] = (yr[i] - mu) * rstd * g[i] + bta[i];
}

// ---------------------------------------------------------------------------
extern "C" void kernel_launch(void* const* d_in, const int* in_sizes, int n_in,
                              void* d_out, int out_size, void* d_ws, size_t ws_size,
                              hipStream_t stream) {
    (void)in_sizes; (void)n_in; (void)out_size; (void)ws_size;
    constexpr int Mr = 2 * 2048, H = 768, FF = 3072, AD = 64;

    const float* hidden = (const float*)d_in[0];
    const float* mask   = (const float*)d_in[1];
    const float* wq = (const float*)d_in[2];  const float* bq = (const float*)d_in[3];
    const float* wk = (const float*)d_in[4];  const float* bk = (const float*)d_in[5];
    const float* wv = (const float*)d_in[6];  const float* bv = (const float*)d_in[7];
    const float* afw = (const float*)d_in[8]; const float* afb = (const float*)d_in[9];
    const float* asw = (const float*)d_in[10];const float* asb = (const float*)d_in[11];
    const float* wi = (const float*)d_in[12]; const float* bi = (const float*)d_in[13];
    const float* wo = (const float*)d_in[14]; const float* bo = (const float*)d_in[15];
    const float* lng = (const float*)d_in[16];const float* lnb = (const float*)d_in[17];

    char* base = (char*)d_ws;
    auto allocF = [&](size_t n) { float* p = (float*)base; base += n * sizeof(float); return p; };
    auto allocH = [&](size_t n) { _Float16* p = (_Float16*)base; base += n * sizeof(_Float16); return p; };

    float* attn   = allocF((size_t)Mr * H);
    float* attn2  = allocF((size_t)Mr * H);
    float* y      = allocF((size_t)Mr * H);
    _Float16* hidH   = allocH((size_t)Mr * H);
    _Float16* wqH    = allocH((size_t)H * H);
    _Float16* wkH    = allocH((size_t)H * H);
    _Float16* wvH    = allocH((size_t)H * H);
    _Float16* afwH   = allocH((size_t)AD * H);
    _Float16* aswH   = allocH((size_t)H * AD);
    _Float16* wiH    = allocH((size_t)FF * H);
    _Float16* woH    = allocH((size_t)H * FF);
    _Float16* qH     = allocH((size_t)Mr * H);
    _Float16* kH     = allocH((size_t)Mr * H);
    _Float16* vH     = allocH((size_t)Mr * H);
    _Float16* attnH  = allocH((size_t)Mr * H);
    _Float16* adpH   = allocH((size_t)Mr * AD);
    _Float16* attn2H = allocH((size_t)Mr * H);
    _Float16* interH = allocH((size_t)Mr * FF);

    dim3 blk(256);
    auto cvt = [&](const float* s, _Float16* d, int n) {
        cvt_kernel<<<dim3((n + 1023) / 1024), blk, 0, stream>>>(s, d, n);
    };
    // one-shot f16 conversion of all GEMM operands
    cvt(hidden, hidH, Mr * H);
    cvt(wq, wqH, H * H);  cvt(wk, wkH, H * H);  cvt(wv, wvH, H * H);
    cvt(afw, afwH, AD * H); cvt(asw, aswH, H * AD);
    cvt(wi, wiH, FF * H); cvt(wo, woH, H * FF);

    // QKV projections (f16 outputs feed attention)
    gemm_f16<<<dim3(H / 64, Mr / 128), blk, 0, stream>>>(hidH, wqH, bq, nullptr, nullptr, qH, Mr, H, H, 0);
    gemm_f16<<<dim3(H / 64, Mr / 128), blk, 0, stream>>>(hidH, wkH, bk, nullptr, nullptr, kH, Mr, H, H, 0);
    gemm_f16<<<dim3(H / 64, Mr / 128), blk, 0, stream>>>(hidH, wvH, bv, nullptr, nullptr, vH, Mr, H, H, 0);
    // attention (f32 for residual, f16 for next GEMM)
    attn_kernel<<<dim3(2048 / 128, 12, 2), blk, 0, stream>>>(qH, kH, vH, mask, attn, attnH);
    // adapter bottleneck (H -> 64 -> H, residual)
    gemm_f16<<<dim3(AD / 64, Mr / 128), blk, 0, stream>>>(attnH, afwH, afb, nullptr, nullptr, adpH, Mr, AD, H, 1);
    gemm_f16<<<dim3(H / 64, Mr / 128),  blk, 0, stream>>>(adpH, aswH, asb, attn, attn2, attn2H, Mr, H, AD, 0);
    // FFN
    gemm_f16<<<dim3(FF / 64, Mr / 128), blk, 0, stream>>>(attn2H, wiH, bi, nullptr, nullptr, interH, Mr, FF, H, 1);
    gemm_f16<<<dim3(H / 64, Mr / 128),  blk, 0, stream>>>(interH, woH, bo, attn2, y, nullptr, Mr, H, FF, 0);
    // final LayerNorm
    ln_kernel<<<dim3(Mr), blk, 0, stream>>>(y, lng, lnb, (float*)d_out);
}
